// Model_7481833030272
// MI455X (gfx1250) — matrix operand order repaired and checked by reading, not compiled, not run on hardware
//
#include <hip/hip_runtime.h>
#include <cstdint>

#define N_    2048
#define T_    600
#define D_    56
#define DK_   30
#define C_    30
#define NH_   4
#define L_    20
#define F_    224
#define HID_  256
#define NL_   (N_*L_)          // 40960
#define KQ_   1680             // C_*D_
#define KTQ_  53               // ceil(1680/32)
#define NTQ_  8                // 128 padded cols (120 real)
#define BNB_  160              // bn stage-1 blocks (160*256 = 40960 rows)

typedef __attribute__((ext_vector_type(16))) __bf16 v16bf;
typedef __attribute__((ext_vector_type(8)))  float  v8f;
typedef int v4i __attribute__((vector_size(16)));

#define AS1 __attribute__((address_space(1)))
#define AS3 __attribute__((address_space(3)))

union ABf { v16bf bf; uint32_t u[8]; };

// native bf16 convert
__device__ inline uint32_t pack2(float a, float b){
  union { uint32_t u; __bf16 h[2]; } x;
  x.h[0] = (__bf16)a; x.h[1] = (__bf16)b;
  return x.u;
}
__device__ inline v8f zero8(){
  v8f z;
#pragma unroll
  for(int i=0;i<8;i++) z[i]=0.f;
  return z;
}
__device__ inline v8f wmma_bf16(const ABf& a, const ABf& b, v8f c){
  return __builtin_amdgcn_wmma_f32_16x16x32_bf16(false, a.bf, false, b.bf,
                                                 (short)0, c, false, false);
}

// ---- A fragment helpers (16x32 bf16 from f32 row-major) -------------------------------
// lane<16: row=lane, K = base+{0..7}, base+{16..23}; lane>=16: row=lane-16, K = base+{8..15}, base+{24..31}
__device__ inline void loadAraw_nc(const float* __restrict__ A, int ld, int row0,
                                   int kbase, int lane, float* ar){
  const float* p = A + (size_t)(row0 + (lane & 15)) * (size_t)ld;
  int kc0 = kbase + ((lane & 16) ? 8 : 0);
#pragma unroll
  for(int i=0;i<8;i++) ar[i]   = p[kc0 + i];
#pragma unroll
  for(int i=0;i<8;i++) ar[8+i] = p[kc0 + 16 + i];
}
// last Q-tile (kt=52): chunk0 (k in [1664..1679]) always valid, chunk1 always OOB -> 0
__device__ inline void loadAraw_qlast(const float* __restrict__ A, int row0,
                                      int lane, float* ar){
  const float* p = A + (size_t)(row0 + (lane & 15)) * (size_t)KQ_;
  int kc0 = 52*32 + ((lane & 16) ? 8 : 0);
#pragma unroll
  for(int i=0;i<8;i++) ar[i]   = p[kc0 + i];
#pragma unroll
  for(int i=0;i<8;i++) ar[8+i] = 0.f;
}
__device__ inline ABf cvtA(const float* ar){
  ABf a;
#pragma unroll
  for(int i=0;i<4;i++){
    a.u[i]   = pack2(ar[2*i],   ar[2*i+1]);
    a.u[4+i] = pack2(ar[8+2*i], ar[8+2*i+1]);
  }
  return a;
}
__device__ inline void loadA_nc(const float* __restrict__ A, int ld, int row0,
                                int kbase, int lane, ABf& a){
  float ar[16];
  loadAraw_nc(A, ld, row0, kbase, lane, ar);
  a = cvtA(ar);
}

// ---- B fragment (32x16 bf16) from packed tile buffer: tile = 32 lanes * 8 dwords ------
__device__ inline void loadB_p(const uint32_t* __restrict__ Bp, int tile, int lane, ABf& b){
  const uint32_t* p = Bp + ((size_t)tile * 32 + lane) * 8;
#pragma unroll
  for(int i=0;i<8;i++) b.u[i] = p[i];
}
__device__ inline void loadB_lds(const uint32_t* p, ABf& b){
#pragma unroll
  for(int i=0;i<8;i++) b.u[i] = p[i];
}

// ---- CDNA5 async global->LDS tile copy (2048 dwords, 128 threads) ---------------------
#if __has_builtin(__builtin_amdgcn_global_load_async_to_lds_b128)
#define HAVE_ASYNC 1
#else
#define HAVE_ASYNC 0
#endif

__device__ inline void copy_tile(const uint32_t* __restrict__ src, uint32_t* dstLds, int tid){
#if HAVE_ASYNC
#pragma unroll
  for(int i=0;i<4;i++)
    __builtin_amdgcn_global_load_async_to_lds_b128(
        (v4i*)(src + i*512 + tid*4),
        (AS3 v4i*)((AS3 uint32_t*)(dstLds + i*512 + tid*4)),
        0, 0);
#else
#pragma unroll
  for(int i=0;i<4;i++){
    uint4 v = *(const uint4*)(src + i*512 + tid*4);
    *(uint4*)(dstLds + i*512 + tid*4) = v;
  }
#endif
}
__device__ inline void wait_tile(){
#if HAVE_ASYNC
#if __has_builtin(__builtin_amdgcn_s_wait_asynccnt)
  __builtin_amdgcn_s_wait_asynccnt(0);
#else
  asm volatile("s_wait_asynccnt 0x0" ::: "memory");
#endif
#endif
}

// ---------------- weight pack kernels --------------------------------------------------
__global__ void pack_wq(const float* __restrict__ wq, uint32_t* __restrict__ Bq){
  int idx = blockIdx.x*256 + threadIdx.x;
  if(idx >= KTQ_*NTQ_*256) return;
  int d    = idx & 7;
  int lane = (idx >> 3) & 31;
  int nt   = (idx >> 8) & 7;
  int kt   = idx >> 11;
  int n  = nt*16 + (lane & 15);
  int k0 = kt*32 + ((lane >> 4) << 3) + ((d & 4) << 2) + 2*(d & 3);
  float f0 = 0.f, f1 = 0.f;
  if(n < NH_*DK_){
    int h = n / DK_, t = n % DK_;
    if(k0   < KQ_){ int c = k0/D_,     dd = k0%D_;     f0 = wq[(((size_t)h*C_+c)*D_+dd)*DK_+t]; }
    if(k0+1 < KQ_){ int c = (k0+1)/D_, dd = (k0+1)%D_; f1 = wq[(((size_t)h*C_+c)*D_+dd)*DK_+t]; }
  }
  Bq[idx] = pack2(f0,f1);
}

__global__ void pack_wk(const float* __restrict__ wk, uint32_t* __restrict__ Bwk){
  int idx = blockIdx.x*256 + threadIdx.x;
  if(idx >= NH_*2*2*256) return;
  int d    = idx & 7;
  int lane = (idx >> 3) & 31;
  int nt   = (idx >> 8) & 1;
  int kt   = (idx >> 9) & 1;
  int h    = idx >> 10;
  int n  = nt*16 + (lane & 15);
  int k0 = kt*32 + ((lane >> 4) << 3) + ((d & 4) << 2) + 2*(d & 3);
  float f0=0.f, f1=0.f;
  if(n < DK_){
    if(k0   < D_) f0 = wk[((size_t)h*D_ + k0  )*DK_ + n];
    if(k0+1 < D_) f1 = wk[((size_t)h*D_ + k0+1)*DK_ + n];
  }
  Bwk[idx] = pack2(f0,f1);
}

__global__ void pack_whh(const float* __restrict__ whh, uint32_t* __restrict__ Bp){
  int idx = blockIdx.x*256 + threadIdx.x;
  if(idx >= 8*48*256) return;
  int d    = idx & 7;
  int lane = (idx >> 3) & 31;
  int nt   = (idx >> 8) % 48;
  int kt   = idx / (48*256);
  int n  = nt*16 + (lane & 15);
  int k0 = kt*32 + ((lane >> 4) << 3) + ((d & 4) << 2) + 2*(d & 3);
  Bp[idx] = pack2(whh[(size_t)n*HID_ + k0], whh[(size_t)n*HID_ + k0 + 1]);
}

__global__ void pack_wih(const float* __restrict__ wih, const float* __restrict__ sfac,
                         uint32_t* __restrict__ Bp){
  int idx = blockIdx.x*256 + threadIdx.x;
  if(idx >= 7*48*256) return;
  int d    = idx & 7;
  int lane = (idx >> 3) & 31;
  int nt   = (idx >> 8) % 48;
  int kt   = idx / (48*256);
  int n  = nt*16 + (lane & 15);
  int k0 = kt*32 + ((lane >> 4) << 3) + ((d & 4) << 2) + 2*(d & 3);
  Bp[idx] = pack2(wih[(size_t)n*F_ + k0]     * sfac[k0],
                  wih[(size_t)n*F_ + k0 + 1] * sfac[k0+1]);
}

// ---------------- Q projection GEMM with async-LDS double-buffered B -------------------
__global__ void __launch_bounds__(128) q_gemm(const float* __restrict__ X,
                                              const uint32_t* __restrict__ Bq,
                                              float* __restrict__ Qbuf){
  __shared__ uint32_t Bs[2][2048];          // 2 x 8KB B tiles
  int tid  = threadIdx.x;
  int lane = tid & 31;
  int wave = tid >> 5;
  int row0 = blockIdx.x*64 + wave*16;

  copy_tile(Bq, &Bs[0][0], tid);            // prefetch tile 0 (async)

  float ar[16];
  loadAraw_nc(X, KQ_, row0, 0, lane, ar);   // prefetch A for kt=0

  v8f acc[NTQ_];
#pragma unroll
  for(int t=0;t<NTQ_;t++) acc[t] = zero8();

  for(int kt=0; kt<KTQ_; kt++){
    int cur = kt & 1;
    wait_tile();                            // my async copies done
    __syncthreads();                        // everyone's copies done / prev reads done
    if(kt+1 < KTQ_)
      copy_tile(Bq + (size_t)(kt+1)*2048, &Bs[cur^1][0], tid);  // prefetch next tile

    ABf a = cvtA(ar);                       // convert current A
    if(kt+1 < KTQ_-1)                       // prefetch next A (full tiles: no guards)
      loadAraw_nc(X, KQ_, row0, (kt+1)*32, lane, ar);
    else if(kt+1 == KTQ_-1)                 // last tile: chunk0 load, chunk1 zero
      loadAraw_qlast(X, row0, lane, ar);

    // one-deep B pipeline: ds-load B[nt+1] while WMMA consumes B[nt]
    ABf bcur, bnxt;
    loadB_lds(&Bs[cur][lane*8], bcur);
#pragma unroll
    for(int nt=0; nt<NTQ_; nt++){
      if(nt+1 < NTQ_) loadB_lds(&Bs[cur][(nt+1)*256 + lane*8], bnxt);
      acc[nt] = wmma_bf16(a, bcur, acc[nt]);
      bcur = bnxt;
    }
  }

  int rbase = row0 + ((lane >> 4) << 3);
#pragma unroll
  for(int nt=0; nt<NTQ_; nt++)
#pragma unroll
    for(int v=0; v<8; v++)
      Qbuf[(size_t)(rbase + v)*128 + nt*16 + (lane & 15)] = acc[nt][v];
}

// ---------------- fused attention: K-proj (WMMA) + QK + softmax + score*x --------------
__global__ void __launch_bounds__(128) attn_kernel(const float* __restrict__ X,
                                                   const float* __restrict__ Qbuf,
                                                   const uint32_t* __restrict__ Bwk,
                                                   float* __restrict__ V){
  __shared__ float xw[32][D_];
  __shared__ float Ks[NH_][32][32];
  __shared__ float sc[NH_][32];
  __shared__ float qs[128];
  int tid  = threadIdx.x;
  int lane = tid & 31;
  int h    = tid >> 5;
  int n = blockIdx.x / L_, l = blockIdx.x % L_;

  const float* xp = X + ((size_t)n*T_ + (size_t)l*C_) * D_;
  for(int i=tid; i<32*D_; i+=128){
    int r = i / D_, d = i % D_;
    xw[r][d] = (r < C_) ? xp[(size_t)r*D_ + d] : 0.f;
  }
  qs[tid] = Qbuf[(size_t)(n*L_ + l)*128 + tid];
  __syncthreads();

  v8f acc[2][2];
#pragma unroll
  for(int mt=0;mt<2;mt++)
#pragma unroll
    for(int nt=0;nt<2;nt++) acc[mt][nt] = zero8();

#pragma unroll
  for(int kt=0; kt<2; kt++){
    ABf a[2], b[2];
#pragma unroll
    for(int mt=0; mt<2; mt++){
      int r   = mt*16 + (lane & 15);
      int kc0 = kt*32 + ((lane & 16) ? 8 : 0);
      int kc1 = kc0 + 16;
#pragma unroll
      for(int i=0;i<4;i++){
        int k0 = kc0 + 2*i, k1 = kc1 + 2*i;
        float f0 = (k0   < D_) ? xw[r][k0]   : 0.f;
        float f1 = (k0+1 < D_) ? xw[r][k0+1] : 0.f;
        float g0 = (k1   < D_) ? xw[r][k1]   : 0.f;
        float g1 = (k1+1 < D_) ? xw[r][k1+1] : 0.f;
        a[mt].u[i]   = pack2(f0,f1);
        a[mt].u[4+i] = pack2(g0,g1);
      }
    }
#pragma unroll
    for(int nt=0; nt<2; nt++) loadB_p(Bwk, ((h*2 + kt)*2 + nt), lane, b[nt]);
#pragma unroll
    for(int nt=0; nt<2; nt++){
      acc[0][nt] = wmma_bf16(a[0], b[nt], acc[0][nt]);
      acc[1][nt] = wmma_bf16(a[1], b[nt], acc[1][nt]);
    }
  }
#pragma unroll
  for(int mt=0;mt<2;mt++)
#pragma unroll
    for(int nt=0;nt<2;nt++)
#pragma unroll
      for(int v=0;v<8;v++)
        Ks[h][mt*16 + ((lane>>4)<<3) + v][nt*16 + (lane & 15)] = acc[mt][nt][v];
  __syncthreads();

  int c = lane;
  float qk = -1e30f;
  if(c < C_){
    float s = 0.f;
#pragma unroll
    for(int t=0;t<DK_;t++) s += Ks[h][c][t] * qs[h*DK_ + t];
    qk = s * 0.18257418583505536f;   // 1/sqrt(30)
  }
  float mx = qk;
#pragma unroll
  for(int o=16;o>0;o>>=1) mx = fmaxf(mx, __shfl_xor(mx, o, 32));
  float e = (c < C_) ? __expf(qk - mx) : 0.f;
  float sm = e;
#pragma unroll
  for(int o=16;o>0;o>>=1) sm += __shfl_xor(sm, o, 32);
  sc[h][lane] = e / sm;
  __syncthreads();

  float* vout = V + ((size_t)l*N_ + n)*F_ + h*D_;
  for(int d=lane; d<D_; d+=32){
    float s = 0.f;
#pragma unroll
    for(int cc=0; cc<C_; cc++) s += sc[h][cc] * xw[cc][d];
    vout[d] = s;
  }
}

// ---------------- BN stats: coalesced two-stage, deterministic -------------------------
__global__ void bn_stats1(const float* __restrict__ V, float* __restrict__ ps,
                          float* __restrict__ pq){
  int f = threadIdx.x;              // 224 threads: thread == feature -> coalesced
  int blk = blockIdx.x;             // 160 blocks x 256 rows
  float s = 0.f, q = 0.f;
  const float* p = V + (size_t)blk*256*F_ + f;
  for(int r=0; r<256; r++){
    float x = p[(size_t)r*F_];
    s += x; q += x*x;
  }
  ps[blk*F_ + f] = s;
  pq[blk*F_ + f] = q;
}

__global__ void bn_stats2(const float* __restrict__ ps, const float* __restrict__ pq,
                          const float* __restrict__ g, const float* __restrict__ b,
                          float* __restrict__ sfac, float* __restrict__ shiftv){
  int f = threadIdx.x;
  float s = 0.f, q = 0.f;
  for(int i=0;i<BNB_;i++){ s += ps[i*F_ + f]; q += pq[i*F_ + f]; }
  float mean = s * (1.f/NL_);
  float var  = q * (1.f/NL_) - mean*mean;
  float sf   = g[f] * rsqrtf(var + 1e-5f);
  sfac[f]   = sf;
  shiftv[f] = b[f] - mean*sf;
}

__global__ void bias_fold(const float* __restrict__ wih, const float* __restrict__ bih,
                          const float* __restrict__ shiftv, float* __restrict__ bgi){
  int n = blockIdx.x*256 + threadIdx.x;
  if(n >= 3*HID_) return;
  float s = bih[n];
  for(int k=0;k<F_;k++) s += wih[(size_t)n*F_ + k] * shiftv[k];
  bgi[n] = s;
}

__global__ void zero_f(float* p, int n){
  int i = blockIdx.x*256 + threadIdx.x;
  if(i < n) p[i] = 0.f;
}

// ---------------- fused GRU step: gi(K=224) + gh(K=256) + gates ------------------------
__global__ void __launch_bounds__(128) gru_step(const float* __restrict__ Vt,
                                                const float* __restrict__ Hin,
                                                const uint32_t* __restrict__ Bwih,
                                                const uint32_t* __restrict__ Bwhh,
                                                const float* __restrict__ bgi,
                                                const float* __restrict__ bhh,
                                                float* __restrict__ Hout){
  int lane = threadIdx.x & 31;
  int wave = threadIdx.x >> 5;
  int row0 = blockIdx.x * 16;
  int j0   = blockIdx.y * 64 + wave * 16;
  int ntr  = j0 >> 4;

  v8f aR = zero8(), aZ = zero8(), aGN = zero8(), aHN = zero8();

  for(int kt=0; kt<7; kt++){                   // gi: K=224 over v (BN folded into Bwih)
    ABf a, b0, b1, b2;
    loadA_nc(Vt, F_, row0, kt*32, lane, a);
    loadB_p(Bwih, kt*48 +      ntr, lane, b0);
    loadB_p(Bwih, kt*48 + 16 + ntr, lane, b1);
    loadB_p(Bwih, kt*48 + 32 + ntr, lane, b2);
    aR  = wmma_bf16(a, b0, aR);
    aZ  = wmma_bf16(a, b1, aZ);
    aGN = wmma_bf16(a, b2, aGN);
  }
  for(int kt=0; kt<8; kt++){                   // gh: K=256 over h
    ABf a, b0, b1, b2;
    loadA_nc(Hin, HID_, row0, kt*32, lane, a);
    loadB_p(Bwhh, kt*48 +      ntr, lane, b0);
    loadB_p(Bwhh, kt*48 + 16 + ntr, lane, b1);
    loadB_p(Bwhh, kt*48 + 32 + ntr, lane, b2);
    aR  = wmma_bf16(a, b0, aR);
    aZ  = wmma_bf16(a, b1, aZ);
    aHN = wmma_bf16(a, b2, aHN);
  }

  int jc  = j0 + (lane & 15);
  float bR  = bgi[jc]          + bhh[jc];
  float bZ  = bgi[HID_ + jc]   + bhh[HID_ + jc];
  float bGN = bgi[2*HID_ + jc];
  float bHN = bhh[2*HID_ + jc];
#pragma unroll
  for(int v=0; v<8; v++){
    int m = row0 + ((lane >> 4) << 3) + v;
    float r  = 1.f / (1.f + __expf(-(aR[v] + bR)));
    float z  = 1.f / (1.f + __expf(-(aZ[v] + bZ)));
    float nn = tanhf(aGN[v] + bGN + r * (aHN[v] + bHN));
    float hp = Hin[(size_t)m*HID_ + jc];
    Hout[(size_t)m*HID_ + jc] = (1.f - z)*nn + z*hp;
  }
}

// ---------------- output linear + sigmoid + BCE (deterministic 2-stage) ----------------
__global__ void out_loss(const float* __restrict__ H, const float* __restrict__ ow,
                         const float* __restrict__ ob, const float* __restrict__ lbl,
                         float* __restrict__ out, float* __restrict__ partial){
  int n = blockIdx.x*256 + threadIdx.x;
  float x = ob[0];
  const float* h = H + (size_t)n*HID_;
  for(int k=0;k<HID_;k++) x += h[k]*ow[k];
  out[1 + n] = 1.f / (1.f + __expf(-x));
  float y = lbl[n];
  float term = fmaxf(x, 0.f) - x*y + log1pf(__expf(-fabsf(x)));
  __shared__ float red[256];
  red[threadIdx.x] = term; __syncthreads();
  for(int s=128; s>0; s>>=1){
    if(threadIdx.x < s) red[threadIdx.x] += red[threadIdx.x + s];
    __syncthreads();
  }
  if(threadIdx.x == 0) partial[blockIdx.x] = red[0];
}

__global__ void finalize(const float* __restrict__ partial, float* __restrict__ out){
  float s = 0.f;
  for(int i=0;i<8;i++) s += partial[i];
  out[0] = s * (1.f / N_);
}

// ====================================================================================
extern "C" void kernel_launch(void* const* d_in, const int* in_sizes, int n_in,
                              void* d_out, int out_size, void* d_ws, size_t ws_size,
                              hipStream_t stream) {
  const float* data  = (const float*)d_in[0];
  const float* lbl   = (const float*)d_in[1];
  const float* wq    = (const float*)d_in[2];
  const float* wk    = (const float*)d_in[3];
  const float* gamma = (const float*)d_in[4];
  const float* beta  = (const float*)d_in[5];
  const float* wih   = (const float*)d_in[6];
  const float* whh   = (const float*)d_in[7];
  const float* bih   = (const float*)d_in[8];
  const float* bhh   = (const float*)d_in[9];
  const float* ow    = (const float*)d_in[10];
  const float* ob    = (const float*)d_in[11];
  float* out = (float*)d_out;

  char* w = (char*)d_ws;
  float* Qbuf   = (float*)w;   w += (size_t)NL_*128*4;
  float* V      = (float*)w;   w += (size_t)NL_*F_*4;
  float* Ha     = (float*)w;   w += (size_t)N_*HID_*4;
  float* Hb     = (float*)w;   w += (size_t)N_*HID_*4;
  float* sfac   = (float*)w;   w += 256*4;
  float* shiftv = (float*)w;   w += 256*4;
  float* bgi    = (float*)w;   w += 768*4;
  float* part   = (float*)w;   w += 64*4;
  float* psum   = (float*)w;   w += (size_t)BNB_*F_*4;
  float* pqsum  = (float*)w;   w += (size_t)BNB_*F_*4;
  uint32_t* Bq   = (uint32_t*)w; w += (size_t)KTQ_*NTQ_*256*4;
  uint32_t* Bwk  = (uint32_t*)w; w += (size_t)NH_*4*256*4;
  uint32_t* Bwhh = (uint32_t*)w; w += (size_t)8*48*256*4;
  uint32_t* Bwih = (uint32_t*)w; w += (size_t)7*48*256*4;

  pack_wq <<<(KTQ_*NTQ_*256 + 255)/256, 256, 0, stream>>>(wq, Bq);
  pack_wk <<<(NH_*4*256 + 255)/256,    256, 0, stream>>>(wk, Bwk);
  pack_whh<<<(8*48*256 + 255)/256,     256, 0, stream>>>(whh, Bwhh);
  zero_f  <<<(N_*HID_ + 255)/256,      256, 0, stream>>>(Ha, N_*HID_);

  q_gemm     <<<NL_/64, 128, 0, stream>>>(data, Bq, Qbuf);
  attn_kernel<<<NL_,    128, 0, stream>>>(data, Qbuf, Bwk, V);

  bn_stats1<<<BNB_, F_, 0, stream>>>(V, psum, pqsum);
  bn_stats2<<<1,    F_, 0, stream>>>(psum, pqsum, gamma, beta, sfac, shiftv);
  pack_wih <<<(7*48*256 + 255)/256, 256, 0, stream>>>(wih, sfac, Bwih);
  bias_fold<<<3, 256, 0, stream>>>(wih, bih, shiftv, bgi);

  for(int t=0; t<L_; t++){
    const float* hin = (t & 1) ? Hb : Ha;
    float*       hou = (t & 1) ? Ha : Hb;
    gru_step<<<dim3(N_/16, HID_/64), 128, 0, stream>>>(
        V + (size_t)t*N_*F_, hin, Bwih, Bwhh, bgi, bhh, hou);
  }
  // after 20 steps (even count), final hidden is in Ha
  out_loss<<<N_/256, 256, 0, stream>>>(Ha, ow, ob, lbl, out, part);
  finalize<<<1, 1, 0, stream>>>(part, out);
}
